// AlignmentModel_7009386627375
// MI455X (gfx1250) — compile-verified, matrix operand order checked by reading, not run on hardware
//
#include <hip/hip_runtime.h>
#include <math.h>

#define N_ 2048
#define L_ 128
#define D_ 512
#define H_ 256
#define B_ 1024
#define EPS_ 0.05f
#define BIG_ 1e4f
#define ITERS_ 50

typedef __bf16 bf16;
typedef __attribute__((ext_vector_type(16))) __bf16 v16bf;
typedef __attribute__((ext_vector_type(8)))  float  v8f;
typedef __attribute__((ext_vector_type(4)))  unsigned int v4u;
typedef __attribute__((ext_vector_type(8)))  int  v8i;
typedef __attribute__((ext_vector_type(4)))  int  v4i;

#if __has_builtin(__builtin_amdgcn_tensor_load_to_lds) && \
    __has_builtin(__builtin_amdgcn_s_wait_tensorcnt)
#define USE_TDM 1
#else
#define USE_TDM 0
#endif

// ---------------------------------------------------------------------------
// WMMA helpers (16x16x32 bf16 -> f32): lanes 0-15 rows / K+0..15, lanes 16-31
// rows / K+16..31, 32B contiguous per lane.  A source row-major [M][K];
// B source must be [N][K] (B-transposed).
// ---------------------------------------------------------------------------
__device__ inline v16bf frag_ld(const bf16* base, int stride, int lane) {
  int r = lane & 15;
  int k = lane & 16;
  return *(const v16bf*)(base + r * stride + k);
}

__device__ inline v8f wmma_bf16(v16bf a, v16bf b, v8f c) {
  return __builtin_amdgcn_wmma_f32_16x16x32_bf16(false, a, false, b,
                                                 (short)0, c, false, false);
}

#if USE_TDM
// ---------------------------------------------------------------------------
// TDM: DMA a [rows][32] bf16 tile (row stride = stride_elems) from global
// into LDS at lds_off (packed [rows][32]).  D# layout per ISA 8.3/8.4:
//   group0: count=1 | lds_addr | global_addr(57b) | type=2
//   group1: data_size=2B, tensor_dim0=stride, tensor_dim1=rows,
//           tile_dim0=32, tile_dim1=rows, tensor_dim0_stride=stride
// Issued by one wave; completion via s_wait_tensorcnt then barrier.
// ---------------------------------------------------------------------------
__device__ __attribute__((always_inline))
void tdm_load_tile32(unsigned lds_off, const void* gptr, int rows, int stride_elems)
{
  unsigned long long ga = (unsigned long long)gptr;
  v4u g0;
  g0.x = 1u;                                   // count=1 (valid user descriptor)
  g0.y = lds_off;                              // LDS byte address
  g0.z = (unsigned)ga;                         // global addr lo
  g0.w = (unsigned)(ga >> 32) | 0x80000000u;   // global addr hi | type=2
  v8i g1;
  unsigned s = (unsigned)stride_elems, rr = (unsigned)rows;
  g1[0] = 0x00010000;                          // data_size = 1 -> 2 bytes
  g1[1] = (int)(s << 16);                      // tensor_dim0 lo16
  g1[2] = (int)((s >> 16) | (rr << 16));       // tensor_dim0 hi / tensor_dim1 lo
  g1[3] = (int)((rr >> 16) | (32u << 16));     // tensor_dim1 hi / tile_dim0=32
  g1[4] = (int)rr;                             // tile_dim1 = rows (tile_dim2=0)
  g1[5] = (int)s;                              // tensor_dim0_stride lo32
  g1[6] = 0;                                   // stride hi / dim1_stride lo
  g1[7] = 0;
  v4i gz = {0, 0, 0, 0};                       // 2D tensor: groups 2/3 zero
#if defined(__clang_major__) && (__clang_major__ >= 23)
  v8i gz8 = {0, 0, 0, 0, 0, 0, 0, 0};
  __builtin_amdgcn_tensor_load_to_lds(g0, g1, gz, gz, gz8, 0);
#else
  __builtin_amdgcn_tensor_load_to_lds(g0, g1, gz, gz, 0);
#endif
}
#endif // USE_TDM

// ---------------------------------------------------------------------------
// K0: convert + transpose weights to bf16 [N][K] so TDM can stage chunks
// ---------------------------------------------------------------------------
__global__ __launch_bounds__(256)
void k_wconv(const float* __restrict__ W, bf16* __restrict__ Wt, int K, int Nn)
{
  long idx = (long)blockIdx.x * 256 + threadIdx.x;
  if (idx < (long)K * Nn) {
    int n = (int)(idx / K);
    int k = (int)(idx % K);
    Wt[idx] = (bf16)W[(long)k * Nn + n];
  }
}

// ---------------------------------------------------------------------------
// K1: embed + 2-layer MLP -> enc bf16 [N*L][H]; one block = 32 tokens.
// ---------------------------------------------------------------------------
__global__ __launch_bounds__(256)
void k_encode(const int* __restrict__ data, const float* __restrict__ emb,
              const bf16* __restrict__ Wa1t, const float* __restrict__ ba1,
              const bf16* __restrict__ Wa2t, const float* __restrict__ ba2,
              bf16* __restrict__ enc)
{
  __shared__ __align__(32) bf16 At[32][512];
  __shared__ __align__(32) bf16 Wt[256][32];
  __shared__ __align__(32) bf16 Hid[32][256];
  const int tid = threadIdx.x, lane = tid & 31, wave = tid >> 5;
  const int tg0 = blockIdx.x * 32;
  const int n = tg0 / L_, t0 = tg0 % L_;
#if USE_TDM
  const unsigned wtoff = (unsigned)(unsigned long long)(void*)&Wt[0][0];
#endif

  { // gather embeddings (fp32 -> bf16)
    int t = tid >> 3;
    int d0 = (tid & 7) * 64;
    int id = data[n * L_ + t0 + t];
    const float* e = emb + (long)id * D_;
#pragma unroll 4
    for (int d = 0; d < 64; ++d) At[t][d0 + d] = (bf16)e[d0 + d];
  }

  const int m0 = (wave & 1) * 16;
  const int nb = (wave >> 1) * 64;

  // ---- layer 1: [32,512] x [512,256]
  v8f acc[4] = {};
  for (int k0 = 0; k0 < D_; k0 += 32) {
    __syncthreads();
#if USE_TDM
    if (tid < 32) {
      tdm_load_tile32(wtoff, Wa1t + k0, 256, D_);
      __builtin_amdgcn_s_wait_tensorcnt(0);
    }
#else
    for (int idx = tid; idx < 256 * 32; idx += 256) {
      int nn = idx & 255, kk = idx >> 8;
      Wt[nn][kk] = Wa1t[(long)nn * D_ + k0 + kk];
    }
#endif
    __syncthreads();
    v16bf a = frag_ld(&At[m0][k0], 512, lane);
#pragma unroll
    for (int tn = 0; tn < 4; ++tn) {
      v16bf bb = frag_ld(&Wt[nb + tn * 16][0], 32, lane);
      acc[tn] = wmma_bf16(a, bb, acc[tn]);
    }
  }
  __syncthreads();
  {
    int colb = nb + (lane & 15);
    int rowb = m0 + ((lane >> 4) << 3);
#pragma unroll
    for (int tn = 0; tn < 4; ++tn) {
      float bias = ba1[colb + tn * 16];
#pragma unroll
      for (int g = 0; g < 8; ++g) {
        float vv = acc[tn][g] + bias;
        Hid[rowb + g][colb + tn * 16] = (bf16)(vv > 0.f ? vv : 0.f);
      }
    }
  }

  // ---- layer 2: [32,256] x [256,256]
  v8f acc2[4] = {};
  for (int k0 = 0; k0 < H_; k0 += 32) {
    __syncthreads();
#if USE_TDM
    if (tid < 32) {
      tdm_load_tile32(wtoff, Wa2t + k0, 256, H_);
      __builtin_amdgcn_s_wait_tensorcnt(0);
    }
#else
    for (int idx = tid; idx < 256 * 32; idx += 256) {
      int nn = idx & 255, kk = idx >> 8;
      Wt[nn][kk] = Wa2t[(long)nn * H_ + k0 + kk];
    }
#endif
    __syncthreads();
    v16bf a = frag_ld(&Hid[m0][k0], 256, lane);
#pragma unroll
    for (int tn = 0; tn < 4; ++tn) {
      v16bf bb = frag_ld(&Wt[nb + tn * 16][0], 32, lane);
      acc2[tn] = wmma_bf16(a, bb, acc2[tn]);
    }
  }
  {
    int colb = nb + (lane & 15);
    int rowb = m0 + ((lane >> 4) << 3);
#pragma unroll
    for (int tn = 0; tn < 4; ++tn) {
      float bias = ba2[colb + tn * 16];
#pragma unroll
      for (int g = 0; g < 8; ++g) {
        float vv = acc2[tn][g] + bias;
        long tok = (long)tg0 + rowb + g;
        enc[tok * H_ + colb + tn * 16] = (bf16)(vv > 0.f ? vv : 0.f);
      }
    }
  }
}

// ---------------------------------------------------------------------------
// K1b: per-token inverse norm
// ---------------------------------------------------------------------------
__global__ __launch_bounds__(256)
void k_invnorm(const bf16* __restrict__ enc, float* __restrict__ invn)
{
  long tok = (long)blockIdx.x * 256 + threadIdx.x;
  const bf16* p = enc + tok * H_;
  float s = 0.f;
  for (int d = 0; d < H_; d += 16) {
    v16bf vv = *(const v16bf*)(p + d);
#pragma unroll
    for (int i = 0; i < 16; ++i) { float f = (float)vv[i]; s += f * f; }
  }
  invn[tok] = 1.0f / (sqrtf(s) + 1e-8f);
}

// ---------------------------------------------------------------------------
// K2: per-pair cost GEMM (TDM-staged WMMA) + 50 Sinkhorn iterations in LDS
// Mb stride 130 (65 dwords) -> conflict-free column traversal.
// ---------------------------------------------------------------------------
__global__ __launch_bounds__(256)
void k_sinkhorn(const int* __restrict__ data, const int* __restrict__ row_idx,
                const int* __restrict__ col_idx, const bf16* __restrict__ enc,
                const float* __restrict__ invn, float* __restrict__ Pout,
                bf16* __restrict__ Pb, bf16* __restrict__ Ptb)
{
  __shared__ bf16 Mb[128][130];
#if USE_TDM
  __shared__ __align__(32) bf16 Ash[128][32];
  __shared__ __align__(32) bf16 Bsh[128][32];
#endif
  __shared__ float u_s[128], v_s[128], loga[128], logb[128];
  __shared__ float invr[128], invc[128];
  __shared__ int rm[128], cm[128];
  __shared__ int cnt[2];
  const int tid = threadIdx.x, lane = tid & 31, wave = tid >> 5;
  const int b = blockIdx.x;
  const int r = row_idx[b], c = col_idx[b];

  if (tid < 2) cnt[tid] = 0;
  __syncthreads();
  if (tid < 128) {
    int mv = (data[r * L_ + tid] != 0);
    rm[tid] = mv; atomicAdd(&cnt[0], mv);
    invr[tid] = invn[(long)r * L_ + tid];
    u_s[tid] = 0.f;
  } else {
    int t = tid - 128;
    int mv = (data[c * L_ + t] != 0);
    cm[t] = mv; atomicAdd(&cnt[1], mv);
    invc[t] = invn[(long)c * L_ + t];
    v_s[t] = 0.f;
  }
  __syncthreads();
  if (tid < 128) loga[tid] = rm[tid] ? -logf((float)cnt[0]) : -1e9f;
  else { int t = tid - 128; logb[t] = cm[t] ? -logf((float)cnt[1]) : -1e9f; }

  // ---- cost GEMM: 128x128x256
  v8f acc[8] = {};
#if USE_TDM
  const bf16* Ab0 = enc + (long)r * L_ * H_;
  const bf16* Bb0 = enc + (long)c * L_ * H_;
  const unsigned aoff = (unsigned)(unsigned long long)(void*)&Ash[0][0];
  const unsigned boff = (unsigned)(unsigned long long)(void*)&Bsh[0][0];
  for (int k0 = 0; k0 < H_; k0 += 32) {
    __syncthreads();
    if (tid < 32) {
      tdm_load_tile32(aoff, Ab0 + k0, 128, H_);
      tdm_load_tile32(boff, Bb0 + k0, 128, H_);
      __builtin_amdgcn_s_wait_tensorcnt(0);
    }
    __syncthreads();
    v16bf a = frag_ld(&Ash[wave * 16][0], 32, lane);
#pragma unroll
    for (int tn = 0; tn < 8; ++tn) {
      v16bf bb = frag_ld(&Bsh[tn * 16][0], 32, lane);
      acc[tn] = wmma_bf16(a, bb, acc[tn]);
    }
  }
#else
  const bf16* Abase = enc + ((long)r * L_ + wave * 16) * H_;
  const bf16* Bbase = enc + (long)c * L_ * H_;
  for (int k0 = 0; k0 < H_; k0 += 32) {
    v16bf a = frag_ld(Abase + k0, H_, lane);
#pragma unroll
    for (int tn = 0; tn < 8; ++tn) {
      v16bf bb = frag_ld(Bbase + (long)tn * 16 * H_ + k0, H_, lane);
      acc[tn] = wmma_bf16(a, bb, acc[tn]);
    }
  }
#endif
  __syncthreads();
  {
    int rowb = wave * 16 + ((lane >> 4) << 3);
    int colb = lane & 15;
#pragma unroll
    for (int tn = 0; tn < 8; ++tn) {
      int col = tn * 16 + colb;
      float ic = invc[col];
      int cmv = cm[col];
#pragma unroll
      for (int g = 0; g < 8; ++g) {
        int row = rowb + g;
        float cost = 1.0f - acc[tn][g] * invr[row] * ic;
        if (!(rm[row] && cmv)) cost = BIG_;
        Mb[row][col] = (bf16)(-cost * (1.0f / EPS_));
      }
    }
  }
  __syncthreads();

  // ---- Sinkhorn: 2 threads per row/col, shfl-combined streaming logsumexp
  const int rc = tid >> 1, half = tid & 1;
  for (int it = 0; it < ITERS_; ++it) {
    {   // u update
      float mx = -3.0e38f;
      for (int j = half * 64; j < half * 64 + 64; ++j)
        mx = fmaxf(mx, (float)Mb[rc][j] + v_s[j]);
      mx = fmaxf(mx, __shfl_xor(mx, 1, 32));
      float s = 0.f;
      for (int j = half * 64; j < half * 64 + 64; ++j)
        s += __expf(((float)Mb[rc][j] + v_s[j]) - mx);
      s += __shfl_xor(s, 1, 32);
      if (half == 0) u_s[rc] = loga[rc] - (mx + __logf(s));
    }
    __syncthreads();
    {   // v update
      float mx = -3.0e38f;
      for (int i = half * 64; i < half * 64 + 64; ++i)
        mx = fmaxf(mx, (float)Mb[i][rc] + u_s[i]);
      mx = fmaxf(mx, __shfl_xor(mx, 1, 32));
      float s = 0.f;
      for (int i = half * 64; i < half * 64 + 64; ++i)
        s += __expf(((float)Mb[i][rc] + u_s[i]) - mx);
      s += __shfl_xor(s, 1, 32);
      if (half == 0) v_s[rc] = logb[rc] - (mx + __logf(s));
    }
    __syncthreads();
  }

  // ---- P = exp(M+u+v)*valid : fp32 -> d_out ; bf16 P / P^T -> workspace
  const long pbase = (long)b * L_ * L_;
  for (int idx = tid; idx < L_ * L_; idx += 256) {
    int i = idx >> 7, j = idx & 127;
    float p = 0.f;
    if (rm[i] && cm[j]) p = __expf((float)Mb[i][j] + u_s[i] + v_s[j]);
    Pout[pbase + idx] = p;
    bf16 pb = (bf16)p;
    Pb[pbase + idx] = pb;
    Ptb[pbase + (long)j * L_ + i] = pb;
  }
}

// ---------------------------------------------------------------------------
// K3: att_col = P @ cols, att_row = P^T @ rows  (per-pair WMMA, 64KB Bt LDS)
// ---------------------------------------------------------------------------
__global__ __launch_bounds__(256)
void k_attend(const int* __restrict__ row_idx, const int* __restrict__ col_idx,
              const bf16* __restrict__ enc, const bf16* __restrict__ Pb,
              const bf16* __restrict__ Ptb, bf16* __restrict__ attc,
              bf16* __restrict__ attr)
{
  __shared__ __align__(32) bf16 Bt[256][128];   // [feature][token]
  const int tid = threadIdx.x, lane = tid & 31, wave = tid >> 5;
  const int b = blockIdx.x;
  const int r = row_idx[b], c = col_idx[b];

  for (int phase = 0; phase < 2; ++phase) {
    const int src = (phase == 0) ? c : r;
    const bf16* A = ((phase == 0) ? Pb : Ptb) + (long)b * L_ * L_;
    bf16* Outp = ((phase == 0) ? attc : attr) + (long)b * L_ * H_;
    __syncthreads();
    for (int idx = tid; idx < H_ * L_; idx += 256) {
      int j = idx >> 8;            // token
      int d = idx & 255;           // feature
      Bt[d][j] = enc[((long)src * L_ + j) * H_ + d];
    }
    __syncthreads();
    const int m0 = wave * 16;
    for (int halfn = 0; halfn < 2; ++halfn) {
      v8f acc[8] = {};
      for (int k0 = 0; k0 < L_; k0 += 32) {
        v16bf a = frag_ld(A + (long)m0 * L_ + k0, L_, lane);
#pragma unroll
        for (int tn = 0; tn < 8; ++tn) {
          v16bf bb = frag_ld(&Bt[halfn * 128 + tn * 16][0] + k0, 128, lane);
          acc[tn] = wmma_bf16(a, bb, acc[tn]);
        }
      }
      int rowb = m0 + ((lane >> 4) << 3);
      int colb = lane & 15;
#pragma unroll
      for (int tn = 0; tn < 8; ++tn) {
        int col = halfn * 128 + tn * 16 + colb;
#pragma unroll
        for (int g = 0; g < 8; ++g)
          Outp[(long)(rowb + g) * H_ + col] = (bf16)acc[tn][g];
      }
    }
  }
}

// ---------------------------------------------------------------------------
// K4: compare MLPs + masked column-sum -> agg [B][512]
// ---------------------------------------------------------------------------
__global__ __launch_bounds__(256)
void k_compare(const int* __restrict__ data, const int* __restrict__ row_idx,
               const int* __restrict__ col_idx, const bf16* __restrict__ enc,
               const bf16* __restrict__ attc, const bf16* __restrict__ attr,
               const bf16* __restrict__ Wc1t, const float* __restrict__ bc1,
               const bf16* __restrict__ Wc2t, const float* __restrict__ bc2,
               float* __restrict__ agg)
{
  __shared__ __align__(32) bf16 Wt[256][32];   // 16KB
  __shared__ __align__(32) bf16 H1[64][256];   // 32KB
  __shared__ float colsum[256];
  __shared__ int msk[128];
  const int tid = threadIdx.x, lane = tid & 31, wave = tid >> 5;
  const int b = blockIdx.x;
  const int m0 = (wave & 3) * 16;
  const int nbase = (wave >> 2) * 128;
#if USE_TDM
  const unsigned wtoff = (unsigned)(unsigned long long)(void*)&Wt[0][0];
#endif

  for (int p = 0; p < 2; ++p) {
    const int seq = (p == 0) ? row_idx[b] : col_idx[b];
    const bf16* X1 = enc + (long)seq * L_ * H_;
    const bf16* X2 = ((p == 0) ? attc : attr) + (long)b * L_ * H_;
    __syncthreads();
    if (tid < 128) msk[tid] = (data[seq * L_ + tid] != 0);
    colsum[tid] = 0.f;
    __syncthreads();

    for (int h = 0; h < 2; ++h) {
      const int roff = h * 64;
      // layer 1: K = 512 (concat via K index)
      v8f acc[8] = {};
      for (int kc = 0; kc < 16; ++kc) {
        int k0 = kc * 32;
        __syncthreads();
#if USE_TDM
        if (tid < 32) {
          tdm_load_tile32(wtoff, Wc1t + k0, 256, 2 * H_);
          __builtin_amdgcn_s_wait_tensorcnt(0);
        }
#else
        for (int idx = tid; idx < 256 * 32; idx += 256) {
          int nn = idx & 255, kk = idx >> 8;
          Wt[nn][kk] = Wc1t[(long)nn * (2 * H_) + k0 + kk];
        }
#endif
        __syncthreads();
        const bf16* Abase = (k0 < H_)
            ? X1 + (long)(roff + m0) * H_ + k0
            : X2 + (long)(roff + m0) * H_ + (k0 - H_);
        v16bf a = frag_ld(Abase, H_, lane);
#pragma unroll
        for (int tn = 0; tn < 8; ++tn) {
          v16bf bb = frag_ld(&Wt[nbase + tn * 16][0], 32, lane);
          acc[tn] = wmma_bf16(a, bb, acc[tn]);
        }
      }
      __syncthreads();
      {
        int rowb = m0 + ((lane >> 4) << 3);
        int colb = lane & 15;
#pragma unroll
        for (int tn = 0; tn < 8; ++tn) {
          int col = nbase + tn * 16 + colb;
          float bias = bc1[col];
#pragma unroll
          for (int g = 0; g < 8; ++g) {
            float vv = acc[tn][g] + bias;
            H1[rowb + g][col] = (bf16)(vv > 0.f ? vv : 0.f);
          }
        }
      }
      // layer 2: K = 256
      v8f acc2[8] = {};
      for (int kc = 0; kc < 8; ++kc) {
        int k0 = kc * 32;
        __syncthreads();
#if USE_TDM
        if (tid < 32) {
          tdm_load_tile32(wtoff, Wc2t + k0, 256, H_);
          __builtin_amdgcn_s_wait_tensorcnt(0);
        }
#else
        for (int idx = tid; idx < 256 * 32; idx += 256) {
          int nn = idx & 255, kk = idx >> 8;
          Wt[nn][kk] = Wc2t[(long)nn * H_ + k0 + kk];
        }
#endif
        __syncthreads();
        v16bf a = frag_ld(&H1[m0][k0], 256, lane);
#pragma unroll
        for (int tn = 0; tn < 8; ++tn) {
          v16bf bb = frag_ld(&Wt[nbase + tn * 16][0], 32, lane);
          acc2[tn] = wmma_bf16(a, bb, acc2[tn]);
        }
      }
      // relu + bias + mask, masked column reduction
      {
        int rowb = m0 + ((lane >> 4) << 3);
        int colb = lane & 15;
#pragma unroll
        for (int tn = 0; tn < 8; ++tn) {
          int col = nbase + tn * 16 + colb;
          float bias = bc2[col];
          float part = 0.f;
#pragma unroll
          for (int g = 0; g < 8; ++g) {
            float vv = acc2[tn][g] + bias;
            vv = vv > 0.f ? vv : 0.f;
            if (msk[roff + rowb + g]) part += vv;
          }
          atomicAdd(&colsum[col], part);
        }
      }
      __syncthreads();
    }
    agg[(long)b * 512 + p * 256 + tid] = colsum[tid];
  }
}

// ---------------------------------------------------------------------------
// K5: tiny head MLP -> logits [B][3]
// ---------------------------------------------------------------------------
__global__ __launch_bounds__(256)
void k_head(const float* __restrict__ agg, const float* __restrict__ Wg1,
            const float* __restrict__ bg1, const float* __restrict__ Wg2,
            const float* __restrict__ bg2, const float* __restrict__ Wo,
            const float* __restrict__ bo, float* __restrict__ logits)
{
  __shared__ float ag[512], h1[256], h2[256];
  const int tid = threadIdx.x;
  const int b = blockIdx.x;
  ag[tid] = agg[(long)b * 512 + tid];
  ag[256 + tid] = agg[(long)b * 512 + 256 + tid];
  __syncthreads();
  float s = bg1[tid];
  for (int k = 0; k < 512; ++k) s += ag[k] * Wg1[k * 256 + tid];
  h1[tid] = s > 0.f ? s : 0.f;
  __syncthreads();
  s = bg2[tid];
  for (int k = 0; k < 256; ++k) s += h1[k] * Wg2[k * 256 + tid];
  h2[tid] = s > 0.f ? s : 0.f;
  __syncthreads();
  if (tid < 3) {
    float o = bo[tid];
    for (int k = 0; k < 256; ++k) o += h2[k] * Wo[k * 3 + tid];
    logits[(long)b * 3 + tid] = o;
  }
}

// ---------------------------------------------------------------------------
extern "C" void kernel_launch(void* const* d_in, const int* in_sizes, int n_in,
                              void* d_out, int out_size, void* d_ws, size_t ws_size,
                              hipStream_t stream)
{
  (void)in_sizes; (void)n_in; (void)out_size; (void)ws_size;
  const int*   data    = (const int*)d_in[0];
  const int*   row_idx = (const int*)d_in[1];
  const int*   col_idx = (const int*)d_in[2];
  const float* emb     = (const float*)d_in[3];
  const float* Wa1 = (const float*)d_in[4];
  const float* ba1 = (const float*)d_in[5];
  const float* Wa2 = (const float*)d_in[6];
  const float* ba2 = (const float*)d_in[7];
  const float* Wc1 = (const float*)d_in[8];
  const float* bc1 = (const float*)d_in[9];
  const float* Wc2 = (const float*)d_in[10];
  const float* bc2 = (const float*)d_in[11];
  const float* Wg1 = (const float*)d_in[12];
  const float* bg1 = (const float*)d_in[13];
  const float* Wg2 = (const float*)d_in[14];
  const float* bg2 = (const float*)d_in[15];
  const float* Wo  = (const float*)d_in[16];
  const float* bo  = (const float*)d_in[17];

  char* w = (char*)d_ws;
  bf16* enc  = (bf16*)w;  w += (size_t)N_ * L_ * H_ * sizeof(bf16);
  float* invn = (float*)w; w += (size_t)N_ * L_ * sizeof(float);
  bf16* Pb   = (bf16*)w;  w += (size_t)B_ * L_ * L_ * sizeof(bf16);
  bf16* Ptb  = (bf16*)w;  w += (size_t)B_ * L_ * L_ * sizeof(bf16);
  bf16* attc = (bf16*)w;  w += (size_t)B_ * L_ * H_ * sizeof(bf16);
  bf16* attr = (bf16*)w;  w += (size_t)B_ * L_ * H_ * sizeof(bf16);
  float* agg = (float*)w; w += (size_t)B_ * 2 * H_ * sizeof(float);
  bf16* Wa1t = (bf16*)w;  w += (size_t)H_ * D_ * sizeof(bf16);
  bf16* Wa2t = (bf16*)w;  w += (size_t)H_ * H_ * sizeof(bf16);
  bf16* Wc1t = (bf16*)w;  w += (size_t)H_ * 2 * H_ * sizeof(bf16);
  bf16* Wc2t = (bf16*)w;  w += (size_t)H_ * H_ * sizeof(bf16);

  float* Pout   = (float*)d_out;
  float* logits = Pout + (size_t)B_ * L_ * L_;

  k_wconv<<<dim3(D_ * H_ / 256), dim3(256), 0, stream>>>(Wa1, Wa1t, D_, H_);
  k_wconv<<<dim3(H_ * H_ / 256), dim3(256), 0, stream>>>(Wa2, Wa2t, H_, H_);
  k_wconv<<<dim3(2 * H_ * H_ / 256), dim3(256), 0, stream>>>(Wc1, Wc1t, 2 * H_, H_);
  k_wconv<<<dim3(H_ * H_ / 256), dim3(256), 0, stream>>>(Wc2, Wc2t, H_, H_);

  k_encode  <<<dim3(N_ * L_ / 32), dim3(256), 0, stream>>>(data, emb, Wa1t, ba1, Wa2t, ba2, enc);
  k_invnorm <<<dim3(N_ * L_ / 256), dim3(256), 0, stream>>>(enc, invn);
  k_sinkhorn<<<dim3(B_), dim3(256), 0, stream>>>(data, row_idx, col_idx, enc, invn, Pout, Pb, Ptb);
  k_attend  <<<dim3(B_), dim3(256), 0, stream>>>(row_idx, col_idx, enc, Pb, Ptb, attc, attr);
  k_compare <<<dim3(B_), dim3(256), 0, stream>>>(data, row_idx, col_idx, enc, attc, attr, Wc1t, bc1, Wc2t, bc2, agg);
  k_head    <<<dim3(B_), dim3(256), 0, stream>>>(agg, Wg1, bg1, Wg2, bg2, Wo, bo, logits);
}
